// HANLayer_12275016532012
// MI455X (gfx1250) — matrix operand order repaired and checked by reading, not compiled, not run on hardware
//
#include <hip/hip_runtime.h>
#include <math.h>

#define NNODES 50000
#define NEDGES 800000
#define INF_   256
#define NHEAD  8
#define HDIM   32
#define HD     256   // NHEAD*HDIM
#define SEMH   128

typedef __attribute__((ext_vector_type(16))) _Float16 v16h;
typedef __attribute__((ext_vector_type(4)))  _Float16 v4h;
typedef __attribute__((ext_vector_type(8)))  float    v8f;

// ---------------- order-preserving float<->uint encode for atomicMax -------
__device__ __forceinline__ unsigned fenc(float f) {
    unsigned b = __float_as_uint(f);
    return b ^ ((unsigned)((int)b >> 31) | 0x80000000u);
}
__device__ __forceinline__ float fdec(unsigned u) {
    unsigned b = (u & 0x80000000u) ? (u ^ 0x80000000u) : ~u;
    return __uint_as_float(b);
}

// ---------------- WMMA GEMM: feat = h @ W  ([N,256]x[256,256], f32 out) ----
#define TILE_M 128
#define TILE_N 64
#define TILE_K 32
#define LDA 40   // padded f16 stride -> conflict-light ds reads
#define LDB 40

__global__ __launch_bounds__(256, 2)
void gemm_feat_kernel(const float* __restrict__ hmat,
                      const float* __restrict__ Wmat,
                      float* __restrict__ feat, int nrows) {
    __shared__ _Float16 As[TILE_M * LDA];
    __shared__ _Float16 Bs[TILE_N * LDB];   // stored [n][k]

    const int tid  = threadIdx.x;
    const int wave = tid >> 5;
    const int lane = tid & 31;
    const int hi   = lane >> 4;     // lane half
    const int lml  = lane & 15;
    const int rowBase = blockIdx.x * TILE_M;
    const int colBase = blockIdx.y * TILE_N;
    const bool fullTile = (rowBase + TILE_M) <= nrows;

    v8f acc[4] = {};   // 4 x (16x16 f32) accumulators -> 16 rows x 64 cols

    for (int kt = 0; kt < INF_; kt += TILE_K) {
        if (fullTile) {
            // A tile 128x32: float4 loads, packed f16x4 LDS stores (b64)
            #pragma unroll
            for (int j = 0; j < 4; ++j) {
                int idx = j * 256 + tid;          // quad index: 128 rows x 8 quads
                int r = idx >> 3, kq = idx & 7;
                const float4 v = *(const float4*)&hmat[(size_t)(rowBase + r) * INF_ + kt + kq * 4];
                v4h hv = { (_Float16)v.x, (_Float16)v.y, (_Float16)v.z, (_Float16)v.w };
                *(v4h*)&As[r * LDA + kq * 4] = hv;
            }
        } else {
            // tail block: guarded scalar staging (runs on 1 of 391 blocks)
            #pragma unroll
            for (int j = 0; j < 16; ++j) {
                int idx = j * 256 + tid;
                int r = idx >> 5, k = idx & 31;
                int grow = rowBase + r;
                float v = (grow < nrows) ? hmat[(size_t)grow * INF_ + kt + k] : 0.0f;
                As[r * LDA + k] = (_Float16)v;
            }
        }
        // B tile 32x64 -> stored transposed Bs[n][k]; float4 loads along n
        #pragma unroll
        for (int j = 0; j < 2; ++j) {
            int idx = j * 256 + tid;              // quad index: 32 k x 16 n-quads
            int nq = idx & 15, kl = idx >> 4;
            const float4 v = *(const float4*)&Wmat[(size_t)(kt + kl) * HD + colBase + nq * 4];
            Bs[(nq * 4 + 0) * LDB + kl] = (_Float16)v.x;
            Bs[(nq * 4 + 1) * LDB + kl] = (_Float16)v.y;
            Bs[(nq * 4 + 2) * LDB + kl] = (_Float16)v.z;
            Bs[(nq * 4 + 3) * LDB + kl] = (_Float16)v.w;
        }
        __syncthreads();

        // A fragment, ISA 16-bit A 16x32 layout (two contiguous 16B runs)
        v16h afrag;
        {
            const _Float16* Ap = &As[(wave * 16 + lml) * LDA];
            #pragma unroll
            for (int vi = 0; vi < 8; ++vi) {
                int kb = ((vi < 4) ? 0 : 16) + hi * 8 + 2 * (vi & 3);
                afrag[2 * vi]     = Ap[kb];
                afrag[2 * vi + 1] = Ap[kb + 1];
            }
        }
        #pragma unroll
        for (int s = 0; s < 4; ++s) {
            // B fragment: col n = s*16+lml, K order same as A fragment (ISA table order)
            v16h bfrag;
            const _Float16* Bp = &Bs[(s * 16 + lml) * LDB];
            #pragma unroll
            for (int vi = 0; vi < 8; ++vi) {
                int kb = ((vi < 4) ? 0 : 16) + hi * 8 + 2 * (vi & 3);
                bfrag[2 * vi]     = Bp[kb];
                bfrag[2 * vi + 1] = Bp[kb + 1];
            }
            acc[s] = __builtin_amdgcn_wmma_f32_16x16x32_f16(
                false, afrag, false, bfrag, (short)0, acc[s], false, false);
        }
        __syncthreads();
    }

    // C layout: VGPR r -> row = r + 8*hi ; col = lml
    #pragma unroll
    for (int s = 0; s < 4; ++s) {
        #pragma unroll
        for (int r = 0; r < 8; ++r) {
            int grow = rowBase + wave * 16 + r + 8 * hi;
            int gcol = colBase + s * 16 + lml;
            if (grow < nrows) feat[(size_t)grow * HD + gcol] = acc[s][r];
        }
    }
}

// ------------- per-node init: el/er dots, zero agg/denom, -inf max ---------
__global__ void node_init_kernel(const float* __restrict__ feat,
                                 const float* __restrict__ attn_l,
                                 const float* __restrict__ attn_r,
                                 float* __restrict__ el, float* __restrict__ er,
                                 unsigned* __restrict__ mEnc,
                                 float* __restrict__ denom,
                                 float* __restrict__ agg) {
    int t = blockIdx.x * blockDim.x + threadIdx.x;
    if (t >= NNODES * NHEAD) return;
    int n = t >> 3, h = t & 7;
    const float* fp = feat + (size_t)n * HD + h * HDIM;
    const float* al = attn_l + h * HDIM;
    const float* ar = attn_r + h * HDIM;
    float* ap = agg + (size_t)n * HD + h * HDIM;
    float sl = 0.f, sr = 0.f;
    #pragma unroll
    for (int d = 0; d < HDIM; ++d) {
        float f = fp[d];
        sl += f * al[d];
        sr += f * ar[d];
        ap[d] = 0.f;
    }
    el[t] = sl;
    er[t] = sr;
    mEnc[t]  = 0x007FFFFFu;   // fenc(-inf)
    denom[t] = 0.f;
}

// ------------- edge pass 1: segment max of leaky-relu scores ---------------
__global__ void edge_max_kernel(const int* __restrict__ src, const int* __restrict__ dst,
                                const float* __restrict__ el, const float* __restrict__ er,
                                unsigned* __restrict__ mEnc) {
    int e = blockIdx.x * blockDim.x + threadIdx.x;
    if (e >= NEDGES) return;
    int s = src[e], d = dst[e];
    const float* elp = el + (size_t)s * NHEAD;
    const float* erp = er + (size_t)d * NHEAD;
    #pragma unroll
    for (int h = 0; h < NHEAD; ++h) {
        float v = elp[h] + erp[h];
        v = (v > 0.f) ? v : 0.2f * v;
        atomicMax(&mEnc[(size_t)d * NHEAD + h], fenc(v));
    }
}

// ------------- edge pass 2: exp, denom, weighted scatter-aggregate ---------
__global__ void edge_scatter_kernel(const int* __restrict__ src, const int* __restrict__ dst,
                                    const float* __restrict__ el, const float* __restrict__ er,
                                    const unsigned* __restrict__ mEnc,
                                    const float* __restrict__ feat,
                                    float* __restrict__ denom, float* __restrict__ agg) {
    int t = blockIdx.x * blockDim.x + threadIdx.x;
    if (t >= NEDGES * NHEAD) return;
    int e = t >> 3, h = t & 7;
    int s = src[e], d = dst[e];
    const float* fp = feat + (size_t)s * HD + h * HDIM;
    float* ap = agg + (size_t)d * HD + h * HDIM;
    // kick the random-gather cachelines early (global_prefetch_b8);
    // overlaps HBM latency with the score/exp math below
    __builtin_prefetch(fp, 0, 0);
    __builtin_prefetch(fp + 16, 0, 0);
    __builtin_prefetch(ap, 1, 0);
    __builtin_prefetch(ap + 16, 1, 0);
    float v = el[(size_t)s * NHEAD + h] + er[(size_t)d * NHEAD + h];
    v = (v > 0.f) ? v : 0.2f * v;
    float m  = fdec(mEnc[(size_t)d * NHEAD + h]);
    float ex = expf(v - m);
    atomicAdd(&denom[(size_t)d * NHEAD + h], ex);
    #pragma unroll
    for (int dd = 0; dd < HDIM; ++dd)
        atomicAdd(&ap[dd], fp[dd] * ex);
}

// ------------- finalize: divide, bias, ELU (in place agg -> z) -------------
__global__ void node_finalize_kernel(float* __restrict__ agg,
                                     const float* __restrict__ denom,
                                     const float* __restrict__ bias) {
    int t = blockIdx.x * blockDim.x + threadIdx.x;
    if (t >= NNODES * HD) return;
    int n = t >> 8, c = t & 255, h = c >> 5;
    float dn = denom[(size_t)n * NHEAD + h];
    float safe = (dn > 0.f) ? dn : 1.0f;
    float v = agg[t] / safe + bias[c];
    agg[t] = (v > 0.f) ? v : expm1f(v);
}

// ------------- semantic attention: s_m = sum_n tanh(z_m W1 + b1) W2 --------
__global__ void sem_zero_kernel(float* sacc) {
    if (threadIdx.x < 2) sacc[threadIdx.x] = 0.f;
}

__global__ void sem_reduce_kernel(const float* __restrict__ z0, const float* __restrict__ z1,
                                  const float* __restrict__ W1, const float* __restrict__ b1,
                                  const float* __restrict__ W2, float* __restrict__ sacc) {
    int gwave  = (blockIdx.x * blockDim.x + threadIdx.x) >> 5;
    int lane   = threadIdx.x & 31;
    int nwaves = (gridDim.x * blockDim.x) >> 5;
    for (int idx = gwave; idx < 2 * NNODES; idx += nwaves) {
        int m = (idx >= NNODES) ? 1 : 0;
        int n = idx - m * NNODES;
        const float* z = (m ? z1 : z0) + (size_t)n * HD;
        float a0 = b1[lane], a1 = b1[lane + 32], a2 = b1[lane + 64], a3 = b1[lane + 96];
        for (int k = 0; k < HD; ++k) {
            float zv = z[k];                       // wave-uniform -> broadcast load
            const float* wr = W1 + (size_t)k * SEMH;
            a0 += zv * wr[lane];
            a1 += zv * wr[lane + 32];
            a2 += zv * wr[lane + 64];
            a3 += zv * wr[lane + 96];
        }
        float p = tanhf(a0) * W2[lane] + tanhf(a1) * W2[lane + 32] +
                  tanhf(a2) * W2[lane + 64] + tanhf(a3) * W2[lane + 96];
        #pragma unroll
        for (int off = 16; off > 0; off >>= 1)
            p += __shfl_xor(p, off, 32);
        if (lane == 0) atomicAdd(&sacc[m], p);
    }
}

// ------------- softmax over 2 metapaths + weighted combine -----------------
__global__ void sem_combine_kernel(const float* __restrict__ z0, const float* __restrict__ z1,
                                   const float* __restrict__ sacc, float* __restrict__ out) {
    int t = blockIdx.x * blockDim.x + threadIdx.x;
    if (t >= NNODES * HD) return;
    float w0 = sacc[0] * (1.0f / NNODES);
    float w1 = sacc[1] * (1.0f / NNODES);
    float mx = fmaxf(w0, w1);
    float e0 = expf(w0 - mx), e1 = expf(w1 - mx);
    float inv = 1.0f / (e0 + e1);
    out[t] = (e0 * inv) * z0[t] + (e1 * inv) * z1[t];
}

// ---------------------------------------------------------------------------
extern "C" void kernel_launch(void* const* d_in, const int* in_sizes, int n_in,
                              void* d_out, int out_size, void* d_ws, size_t ws_size,
                              hipStream_t stream) {
    const float* h       = (const float*)d_in[0];
    const int*   src_a   = (const int*)  d_in[1];
    const int*   dst_a   = (const int*)  d_in[2];
    const int*   src_b   = (const int*)  d_in[3];
    const int*   dst_b   = (const int*)  d_in[4];
    const float* W_a     = (const float*)d_in[5];
    const float* al_a    = (const float*)d_in[6];
    const float* ar_a    = (const float*)d_in[7];
    const float* bias_a  = (const float*)d_in[8];
    const float* W_b     = (const float*)d_in[9];
    const float* al_b    = (const float*)d_in[10];
    const float* ar_b    = (const float*)d_in[11];
    const float* bias_b  = (const float*)d_in[12];
    const float* sem_W1  = (const float*)d_in[13];
    const float* sem_b1  = (const float*)d_in[14];
    const float* sem_W2  = (const float*)d_in[15];
    float* out = (float*)d_out;

    // workspace layout (floats)
    float* ws = (float*)d_ws;
    const size_t NHD = (size_t)NNODES * HD;     // 12.8M floats
    const size_t NH  = (size_t)NNODES * NHEAD;  // 400k floats
    float*    z0    = ws;                 // N*HD   (also agg for metapath a)
    float*    z1    = z0 + NHD;           // N*HD   (also agg for metapath b)
    float*    feat  = z1 + NHD;           // N*HD   (reused per metapath)
    float*    el    = feat + NHD;         // N*H
    float*    er    = el + NH;            // N*H
    unsigned* mEnc  = (unsigned*)(er + NH); // N*H
    float*    denom = (float*)(mEnc + NH);  // N*H
    float*    sacc  = denom + NH;           // 2

    dim3 gGrid((NNODES + TILE_M - 1) / TILE_M, HD / TILE_N);
    int  nhBlocks  = (NNODES * NHEAD + 255) / 256;
    int  eBlocks   = (NEDGES + 255) / 256;
    int  ehBlocks  = (NEDGES * NHEAD + 255) / 256;
    int  nhdBlocks = (NNODES * HD + 255) / 256;

    // ---- metapath A ----
    gemm_feat_kernel<<<gGrid, 256, 0, stream>>>(h, W_a, feat, NNODES);
    node_init_kernel<<<nhBlocks, 256, 0, stream>>>(feat, al_a, ar_a, el, er, mEnc, denom, z0);
    edge_max_kernel<<<eBlocks, 256, 0, stream>>>(src_a, dst_a, el, er, mEnc);
    edge_scatter_kernel<<<ehBlocks, 256, 0, stream>>>(src_a, dst_a, el, er, mEnc, feat, denom, z0);
    node_finalize_kernel<<<nhdBlocks, 256, 0, stream>>>(z0, denom, bias_a);

    // ---- metapath B (reuses feat/el/er/mEnc/denom) ----
    gemm_feat_kernel<<<gGrid, 256, 0, stream>>>(h, W_b, feat, NNODES);
    node_init_kernel<<<nhBlocks, 256, 0, stream>>>(feat, al_b, ar_b, el, er, mEnc, denom, z1);
    edge_max_kernel<<<eBlocks, 256, 0, stream>>>(src_b, dst_b, el, er, mEnc);
    edge_scatter_kernel<<<ehBlocks, 256, 0, stream>>>(src_b, dst_b, el, er, mEnc, feat, denom, z1);
    node_finalize_kernel<<<nhdBlocks, 256, 0, stream>>>(z1, denom, bias_b);

    // ---- semantic attention ----
    sem_zero_kernel<<<1, 32, 0, stream>>>(sacc);
    // 12500 blocks * 8 waves = 100000 waves == 2*N node tasks
    sem_reduce_kernel<<<12500, 256, 0, stream>>>(z0, z1, sem_W1, sem_b1, sem_W2, sacc);
    sem_combine_kernel<<<nhdBlocks, 256, 0, stream>>>(z0, z1, sacc, out);

    (void)in_sizes; (void)n_in; (void)out_size; (void)ws_size;
}
